// RNNEncDec_57320633533038
// MI455X (gfx1250) — compile-verified, matrix operand order checked by reading, not run on hardware
//
#include <hip/hip_runtime.h>
#include <hip/hip_bf16.h>
#include <math.h>

// ---------------------------------------------------------------------------
// RNN encoder-decoder for MI455X (gfx1250).
// Dominant cost: 127 x (256 x 32000 x 300) logits GEMM -> bf16 WMMA, f32 acc.
// Logits kernel: 64 batch rows per block (A staged in LDS via async-to-LDS),
// cutting outW L2 traffic 4x vs a 16-row tiling.
// ---------------------------------------------------------------------------

typedef __attribute__((ext_vector_type(16))) __bf16 v16bf;
typedef __attribute__((ext_vector_type(8)))  __bf16 v8bf;
typedef __attribute__((ext_vector_type(8)))  float  v8f;
typedef __attribute__((ext_vector_type(4)))  int    v4i;

#define HH     300   // hidden
#define HP     304   // hidden padded to 16 (N dim of recurrence GEMM)
#define KP     320   // K padded to 32 (10 WMMA k-steps)
#define BB     256   // batch
#define VV     32000 // vocab
#define NXS    128
#define NYS    128
#define MT     4     // M tiles per logits block (64 batch rows)
#define NCH    (VV / 128)  // 250 vocab chunks per step (block N span = 8 waves x 16)

#if defined(__has_builtin)
#if __has_builtin(__builtin_amdgcn_global_load_async_to_lds_b128) && \
    __has_builtin(__builtin_amdgcn_s_wait_asynccnt)
#define HAVE_ASYNC_LDS 1
#endif
#endif
#ifndef HAVE_ASYNC_LDS
#define HAVE_ASYNC_LDS 0
#endif

#if HAVE_ASYNC_LDS
// 16B global -> LDS async copy (GLOBAL_LOAD_ASYNC_TO_LDS_B128, ASYNCcnt).
static __device__ inline void async_copy16(const void* g, void* l) {
    __builtin_amdgcn_global_load_async_to_lds_b128(
        (__attribute__((address_space(1))) v4i*)g,
        (__attribute__((address_space(3))) v4i*)l,
        0, 0);
}
#endif

// Load one A/B fragment for V_WMMA_*_16x16x32_bf16 from a row-major bf16
// buffer with row stride KP.
// Lanes 0-15: row = base_row + (lane&15), K = {k0..k0+7, k0+16..k0+23}
// Lanes16-31: same rows,                 K = {k0+8..k0+15, k0+24..k0+31}
static __device__ inline v16bf load_frag(const __bf16* base,
                                         int row, int k0, int lane) {
    const __bf16* p = base + (long)row * KP + k0 + ((lane & 16) ? 8 : 0);
    v8bf lo = *(const v8bf*)(p);        // elements 0..7  -> VGPR0..3
    v8bf hi = *(const v8bf*)(p + 16);   // elements 8..15 -> VGPR4..7
    return __builtin_shufflevector(lo, hi, 0, 1, 2, 3, 4, 5, 6, 7,
                                           8, 9, 10, 11, 12, 13, 14, 15);
}

static __device__ inline void sm_merge(float& m, float& s, float m2, float s2) {
    float nm = fmaxf(m, m2);
    s = s * __expf(m - nm) + s2 * __expf(m2 - nm);
    m = nm;
}

// ---------------------------------------------------------------------------
// Convert f32 [rows x 300] -> bf16 [prows x 320], zero padded.
// ---------------------------------------------------------------------------
__global__ void convert_pad_kernel(const float* __restrict__ in,
                                   __bf16* __restrict__ out,
                                   int rows, int prows) {
    long i = (long)blockIdx.x * blockDim.x + threadIdx.x;
    long total = (long)prows * KP;
    if (i >= total) return;
    int r = (int)(i / KP);
    int c = (int)(i % KP);
    float v = (r < rows && c < HH) ? in[(long)r * HH + c] : 0.0f;
    out[i] = (__bf16)v;
}

// ---------------------------------------------------------------------------
// One recurrence step: hnew = relu(emb[idx] + h @ W^T + b)
// grid = 38 blocks x 256 thr (8 waves); one 16x16 tile per wave; 304 tiles.
// ---------------------------------------------------------------------------
__global__ void rnn_step_kernel(const __bf16* __restrict__ hbf_in,  // [BB*KP]
                                const __bf16* __restrict__ Wbf,     // [HP*KP]
                                const float*  __restrict__ bias,    // [HH]
                                const float*  __restrict__ emb,     // [V*HH]
                                const int*    __restrict__ idx,     // [BB]
                                float*        __restrict__ hf_out,  // [BB*KP]
                                __bf16*       __restrict__ hbf_out) // [BB*KP]
{
    const int lane = threadIdx.x & 31;
    const int tile = blockIdx.x * 8 + (threadIdx.x >> 5);
    if (tile >= 16 * (HP / 16)) return;
    const int m0 = (tile / (HP / 16)) * 16;   // batch tile
    const int n0 = (tile % (HP / 16)) * 16;   // hidden tile
    const int r  = lane & 15;

    v8f acc = {};
#pragma unroll
    for (int ks = 0; ks < KP / 32; ++ks) {
        v16bf a = load_frag(hbf_in, m0 + r, ks * 32, lane);
        v16bf b = load_frag(Wbf,    n0 + r, ks * 32, lane);
        acc = __builtin_amdgcn_wmma_f32_16x16x32_bf16(
            false, a, false, b, (short)0, acc, false, false);
    }

    // Epilogue: C layout -> lane&15 = N col, VGPR v = row v (+8 for hi half)
    const int j  = n0 + r;
    const int hi = (lane >> 4) & 1;
#pragma unroll
    for (int v = 0; v < 8; ++v) {
        const int brow = m0 + v + 8 * hi;
        float val = 0.0f;
        if (j < HH) {
            const int tok = idx[brow];
            val = acc[v] + bias[j] + emb[(long)tok * HH + j];
            val = fmaxf(val, 0.0f);
        }
        hf_out[(long)brow * KP + j]  = val;
        hbf_out[(long)brow * KP + j] = (__bf16)val;
    }
}

// ---------------------------------------------------------------------------
// Logits GEMM + split online softmax.
// grid = (NCH, 4), 256 thr. Block owns 64 batch rows x 128 vocab cols.
// A tile (64 x 320 bf16 = 40KB) staged in LDS (async-to-LDS when available);
// each wave owns one 16-col N tile and runs 4 WMMAs per k-step sharing the
// B fragment. Writes (max, sumexp) per (batch row, chunk).
// ---------------------------------------------------------------------------
__global__ void logits_softmax_kernel(const __bf16* __restrict__ hbf,   // [BB*KP]
                                      const __bf16* __restrict__ Wbf,   // [VV*KP]
                                      const float*  __restrict__ outb,  // [VV]
                                      float2*       __restrict__ partials) // [BB*NCH]
{
    __shared__ __align__(16) __bf16 Atile[64 * KP];   // 40 KB
    __shared__ float lm[64][8];
    __shared__ float ls[64][8];

    const int lane  = threadIdx.x & 31;
    const int wave  = threadIdx.x >> 5;
    const int m0    = blockIdx.y * 64;
    const int chunk = blockIdx.x;
    const int nbase = chunk * 128 + wave * 16;
    const int r     = lane & 15;
    const int hi    = (lane >> 4) & 1;

    // Stage A tile: 64 rows x 320 bf16 = 40960 B = 2560 x 16B chunks.
    {
        const char* gsrc = (const char*)(hbf + (long)m0 * KP);
        char*       ldst = (char*)Atile;
#pragma unroll
        for (int i = 0; i < 10; ++i) {
            const int c = (threadIdx.x + 256 * i) * 16;
#if HAVE_ASYNC_LDS
            async_copy16(gsrc + c, ldst + c);
#else
            *(v8bf*)(ldst + c) = *(const v8bf*)(gsrc + c);
#endif
        }
#if HAVE_ASYNC_LDS
        __builtin_amdgcn_s_wait_asynccnt(0);
#endif
        __syncthreads();
    }

    v8f acc[MT];
#pragma unroll
    for (int mt = 0; mt < MT; ++mt) acc[mt] = (v8f){};

    for (int ks = 0; ks < KP / 32; ++ks) {
        v16bf a[MT];
#pragma unroll
        for (int mt = 0; mt < MT; ++mt)
            a[mt] = load_frag(Atile, mt * 16 + r, ks * 32, lane);
        v16bf b = load_frag(Wbf, nbase + r, ks * 32, lane);
#pragma unroll
        for (int mt = 0; mt < MT; ++mt)
            acc[mt] = __builtin_amdgcn_wmma_f32_16x16x32_bf16(
                false, a[mt], false, b, (short)0, acc[mt], false, false);
    }

    // Per-row (max, sumexp) over this wave's 16 columns; combine waves in LDS.
    const float bias = outb[nbase + r];
#pragma unroll
    for (int mt = 0; mt < MT; ++mt) {
#pragma unroll
        for (int v = 0; v < 8; ++v) {
            float x = acc[mt][v] + bias;
            float m = x;
#pragma unroll
            for (int off = 8; off >= 1; off >>= 1)
                m = fmaxf(m, __shfl_xor(m, off, 16));
            float e = __expf(x - m);
#pragma unroll
            for (int off = 8; off >= 1; off >>= 1)
                e += __shfl_xor(e, off, 16);
            if ((lane & 15) == 0) {
                const int row = mt * 16 + v + 8 * hi;
                lm[row][wave] = m;
                ls[row][wave] = e;
            }
        }
    }
    __syncthreads();
    if (threadIdx.x < 64) {
        const int row = threadIdx.x;
        float m = -1e30f, s = 0.0f;
#pragma unroll
        for (int w = 0; w < 8; ++w) sm_merge(m, s, lm[row][w], ls[row][w]);
        partials[(long)(m0 + row) * NCH + chunk] = make_float2(m, s);
    }
}

// ---------------------------------------------------------------------------
// Per-step CE: combine chunk partials -> logsumexp; exact f32 target logit;
// out[0] += mean_b(lse - target_logit).   1 block x 256 threads.
// ---------------------------------------------------------------------------
__global__ void ce_reduce_kernel(const float2* __restrict__ partials,
                                 const float*  __restrict__ hf,    // [BB*KP]
                                 const float*  __restrict__ outW,  // [VV*HH]
                                 const float*  __restrict__ outb,  // [VV]
                                 const int*    __restrict__ tgt,   // [BB]
                                 float*        __restrict__ out)
{
    const int b = threadIdx.x;
    float m = -1e30f, s = 0.0f;
    for (int c = 0; c < NCH; ++c) {
        float2 p = partials[(long)b * NCH + c];
        sm_merge(m, s, p.x, p.y);
    }
    const float lse = m + __logf(s);

    const int t = tgt[b];
    const float* hr = hf + (long)b * KP;
    const float* wr = outW + (long)t * HH;
    float dot = outb[t];
    for (int k = 0; k < HH; ++k) dot = fmaf(hr[k], wr[k], dot);

    __shared__ float red[BB];
    red[b] = lse - dot;
    __syncthreads();
    for (int off = BB / 2; off > 0; off >>= 1) {
        if (b < off) red[b] += red[b + off];
        __syncthreads();
    }
    if (b == 0) out[0] += red[0] * (1.0f / (float)BB);
}

// ---------------------------------------------------------------------------
extern "C" void kernel_launch(void* const* d_in, const int* in_sizes, int n_in,
                              void* d_out, int out_size, void* d_ws, size_t ws_size,
                              hipStream_t stream) {
    const int*   x       = (const int*)d_in[0];
    const int*   y       = (const int*)d_in[1];
    const float* enc_emb = (const float*)d_in[2];
    const float* encW    = (const float*)d_in[3];
    const float* encb    = (const float*)d_in[4];
    const float* dec_emb = (const float*)d_in[5];
    const float* decW    = (const float*)d_in[6];
    const float* decb    = (const float*)d_in[7];
    const float* outW    = (const float*)d_in[8];
    const float* outb    = (const float*)d_in[9];
    float* out = (float*)d_out;

    // Workspace carve-up (256B aligned).
    char* ws = (char*)d_ws;
    size_t off = 0;
    auto carve = [&](size_t bytes) -> char* {
        char* p = ws + off;
        off = (off + bytes + 255) & ~(size_t)255;
        return p;
    };
    __bf16* outWbf  = (__bf16*)carve((size_t)VV * KP * sizeof(__bf16));
    __bf16* encWbf  = (__bf16*)carve((size_t)HP * KP * sizeof(__bf16));
    __bf16* decWbf  = (__bf16*)carve((size_t)HP * KP * sizeof(__bf16));
    __bf16* hbf[2]; float* hf[2];
    hbf[0] = (__bf16*)carve((size_t)BB * KP * sizeof(__bf16));
    hbf[1] = (__bf16*)carve((size_t)BB * KP * sizeof(__bf16));
    hf[0]  = (float*)carve((size_t)BB * KP * sizeof(float));
    hf[1]  = (float*)carve((size_t)BB * KP * sizeof(float));
    float2* partials = (float2*)carve((size_t)BB * NCH * sizeof(float2));

    (void)hipMemsetAsync(out, 0, sizeof(float), stream);
    (void)hipMemsetAsync(hbf[0], 0, (size_t)BB * KP * sizeof(__bf16), stream);
    (void)hipMemsetAsync(hbf[1], 0, (size_t)BB * KP * sizeof(__bf16), stream);

    // One-time weight conversions to padded bf16.
    {
        long n = (long)HP * KP;
        convert_pad_kernel<<<(unsigned)((n + 255) / 256), 256, 0, stream>>>(
            encW, encWbf, HH, HP);
        convert_pad_kernel<<<(unsigned)((n + 255) / 256), 256, 0, stream>>>(
            decW, decWbf, HH, HP);
        long nv = (long)VV * KP;
        convert_pad_kernel<<<(unsigned)((nv + 255) / 256), 256, 0, stream>>>(
            outW, outWbf, VV, VV);
    }

    const int rnn_blocks = (16 * (HP / 16) + 7) / 8;  // 38

    // Encoder: 128 steps, h double-buffered (step t reads buf t&1, writes (t+1)&1).
    for (int t = 0; t < NXS; ++t) {
        rnn_step_kernel<<<rnn_blocks, 256, 0, stream>>>(
            hbf[t & 1], encWbf, encb, enc_emb, x + (long)t * BB,
            hf[(t + 1) & 1], hbf[(t + 1) & 1]);
    }
    // Decoder: 127 teacher-forced steps; encoder left final h in buf 0.
    for (int t = 0; t < NYS - 1; ++t) {
        rnn_step_kernel<<<rnn_blocks, 256, 0, stream>>>(
            hbf[t & 1], decWbf, decb, dec_emb, y + (long)t * BB,
            hf[(t + 1) & 1], hbf[(t + 1) & 1]);
        logits_softmax_kernel<<<dim3(NCH, 4), 256, 0, stream>>>(
            hbf[(t + 1) & 1], outWbf, outb, partials);
        ce_reduce_kernel<<<1, BB, 0, stream>>>(
            partials, hf[(t + 1) & 1], outW, outb, y + (long)(t + 1) * BB, out);
    }
}
